// MultiHeadDotProductAttention_51745765982874
// MI455X (gfx1250) — compile-verified, matrix operand order checked by reading, not run on hardware
//
#include <hip/hip_runtime.h>
#include <hip/hip_bf16.h>

// ---------------- problem constants ----------------
constexpr int Bc = 4;
constexpr int Sc = 2048;
constexpr int Dc = 1024;               // model dim
constexpr int Hc = 16;
constexpr int DHc = 64;
constexpr int HDc = Hc * DHc;          // 1024
constexpr int Mrows = Bc * Sc;         // 8192

typedef unsigned short u16;
typedef __attribute__((ext_vector_type(8)))  u16    u16x8;
typedef __attribute__((ext_vector_type(16))) __bf16 v16bf;
typedef __attribute__((ext_vector_type(8)))  float  v8f;

union Frag16 {
    u16x8 u8[2];
    v16bf bf;
};

__device__ __forceinline__ u16 f2bf(float x) {
    unsigned u = __float_as_uint(x);
    unsigned r = u + 0x7FFFu + ((u >> 16) & 1u);   // round-to-nearest-even
    return (u16)(r >> 16);
}

__device__ __forceinline__ v8f wmma_bf16(v16bf a, v16bf b, v8f c) {
    return __builtin_amdgcn_wmma_f32_16x16x32_bf16(
        false, a, false, b, (short)0, c, false, false);
}

#if __has_builtin(__builtin_amdgcn_exp2f)
#define EXP2F(x) __builtin_amdgcn_exp2f(x)
#else
#define EXP2F(x) exp2f(x)
#endif

// ---------------- Tensor Data Mover (CDNA5 TDM) ----------------
#if __has_builtin(__builtin_amdgcn_tensor_load_to_lds)
#define HAVE_TDM 1
typedef __attribute__((ext_vector_type(4))) unsigned int u32x4;
typedef __attribute__((ext_vector_type(8))) int          i32x8;
typedef __attribute__((ext_vector_type(4))) int          i32x4;

// 2D tile load: rows x cols (bf16 elements), global row stride (elements),
// LDS destination padded by pad_amount DWORDs every 2^(pad_interval+1) DWORDs.
__device__ __forceinline__ void tdm_load_2d(const u16* gptr, const void* lptr,
                                            unsigned rows, unsigned cols,
                                            unsigned row_stride,
                                            unsigned pad_interval_code,
                                            unsigned pad_amount_code) {
    // LDS aperture keeps the LDS byte offset in addr[31:0] (ISA 10.2)
    unsigned lds = (unsigned)(unsigned long long)lptr;
    unsigned long long ga = (unsigned long long)gptr;
    u32x4 g0;
    g0[0] = 1u;                                               // count=1, user mode
    g0[1] = lds;                                              // lds_addr
    g0[2] = (unsigned)ga;                                     // global_addr[31:0]
    g0[3] = (unsigned)((ga >> 32) & 0x1FFFFFFull) | (2u << 30); // addr[56:32] | type=2
    i32x8 g1;
    g1[0] = (int)((1u << 16)                                  // data_size = 2 bytes
                | (1u << 20)                                  // pad_enable
                | (pad_interval_code << 22)
                | (pad_amount_code << 25));
    g1[1] = (int)((cols & 0xFFFFu) << 16);                    // tensor_dim0[15:0]
    g1[2] = (int)((rows & 0xFFFFu) << 16);                    // dim0 hi=0 | tensor_dim1 lo
    g1[3] = (int)((cols & 0xFFFFu) << 16);                    // dim1 hi=0 | tile_dim0
    g1[4] = (int)(rows & 0xFFFFu);                            // tile_dim1 | tile_dim2=0
    g1[5] = (int)row_stride;                                  // tensor_dim0_stride[31:0]
    g1[6] = 0;
    g1[7] = 0;
    i32x4 z4 = {0, 0, 0, 0};
#if __clang_major__ >= 23
    i32x8 z8 = {0, 0, 0, 0, 0, 0, 0, 0};
    __builtin_amdgcn_tensor_load_to_lds(g0, g1, z4, z4, z8, 0);
#else
    __builtin_amdgcn_tensor_load_to_lds(g0, g1, z4, z4, 0);
#endif
}
#endif

// ---------------- fp32 -> bf16 cast ----------------
__global__ __launch_bounds__(256) void cast_f32_bf16(const float* __restrict__ in,
                                                     u16* __restrict__ out, int n4) {
    int i = blockIdx.x * blockDim.x + threadIdx.x;
    if (i < n4) {
        float4 f = ((const float4*)in)[i];
        ushort4 o;
        o.x = f2bf(f.x); o.y = f2bf(f.y); o.z = f2bf(f.z); o.w = f2bf(f.w);
        ((ushort4*)out)[i] = o;
    }
}

// ---------------- fp32 -> bf16 cast + transpose (1024x1024 weights) ----------------
__global__ __launch_bounds__(256) void cast_transpose(const float* __restrict__ in,
                                                      u16* __restrict__ out) {
    __shared__ u16 tile[32][33];
    const int x0 = blockIdx.x * 32, y0 = blockIdx.y * 32;
    const int tx = threadIdx.x & 31, ty = threadIdx.x >> 5;   // 32 x 8
    #pragma unroll
    for (int r = 0; r < 32; r += 8)
        tile[ty + r][tx] = f2bf(in[(size_t)(y0 + ty + r) * 1024 + x0 + tx]);
    __syncthreads();
    #pragma unroll
    for (int r = 0; r < 32; r += 8)
        out[(size_t)(x0 + ty + r) * 1024 + y0 + tx] = tile[tx][ty + r];
}

// ---------------- bf16 GEMM: C[8192x1024] = A[8192x1024] * WT^T ----------------
// WT is the pre-transposed weight [n][k]. OUT_MODE: 0=f32 [m][n], 1=bf16 [m][n],
// 2=bf16 in attention-V layout Vt[(b*1024+n)*2048 + s].
template <int OUT_MODE>
__global__ __launch_bounds__(256) void gemm_kernel(const u16* __restrict__ A,
                                                   const u16* __restrict__ WT,
                                                   const float* __restrict__ bias,
                                                   void* __restrict__ out,
                                                   float scale) {
    constexpr int Kdim = 1024, Ndim = 1024;
    __shared__ u16 lA[2][128][40];   // [m][k], pitch 80B (TDM-padded)
    __shared__ u16 lB[2][128][40];   // [n][k]

    const int tid = threadIdx.x;
    const int wid = tid >> 5;
    const int lane = tid & 31;
    const int lr = lane & 15;
    const int hi = lane >> 4;

    const int n0 = blockIdx.x * 128;
    const int m0 = blockIdx.y * 128;
    const int wm = wid & 3;          // 4 waves along M
    const int wn = wid >> 2;         // 2 waves along N

    const u16* Abase = A  + (size_t)m0 * Kdim;
    const u16* Bbase = WT + (size_t)n0 * Kdim;

    v8f acc[2][4] = {};

#ifdef HAVE_TDM
    if (wid == 0) {
        tdm_load_2d(Abase, &lA[0][0][0], 128, 32, Kdim, /*16 dw*/3, /*4 dw*/3);
        tdm_load_2d(Bbase, &lB[0][0][0], 128, 32, Kdim, 3, 3);
    }
#endif

    for (int k0 = 0; k0 < Kdim; k0 += 32) {
        const int cur = (k0 >> 5) & 1;
#ifdef HAVE_TDM
        if (wid == 0) {
            const int nxt = cur ^ 1;
            if (k0 + 32 < Kdim) {
                tdm_load_2d(Abase + k0 + 32, &lA[nxt][0][0], 128, 32, Kdim, 3, 3);
                tdm_load_2d(Bbase + k0 + 32, &lB[nxt][0][0], 128, 32, Kdim, 3, 3);
                __builtin_amdgcn_s_wait_tensorcnt((short)2);
            } else {
                __builtin_amdgcn_s_wait_tensorcnt((short)0);
            }
        }
#else
        {
            const int row = tid >> 1;
            const int col = (tid & 1) * 16;
            const u16x8* as = (const u16x8*)(Abase + (size_t)row * Kdim + k0 + col);
            *(u16x8*)&lA[cur][row][col]     = as[0];
            *(u16x8*)&lA[cur][row][col + 8] = as[1];
            const u16x8* bs = (const u16x8*)(Bbase + (size_t)row * Kdim + k0 + col);
            *(u16x8*)&lB[cur][row][col]     = bs[0];
            *(u16x8*)&lB[cur][row][col + 8] = bs[1];
        }
#endif
        __syncthreads();

        Frag16 af[2], bf[4];
        #pragma unroll
        for (int mt = 0; mt < 2; ++mt) {
            const int row = wm * 32 + mt * 16 + lr;
            af[mt].u8[0] = *(const u16x8*)&lA[cur][row][hi * 8];
            af[mt].u8[1] = *(const u16x8*)&lA[cur][row][16 + hi * 8];
        }
        #pragma unroll
        for (int nt = 0; nt < 4; ++nt) {
            const int col = wn * 64 + nt * 16 + lr;
            bf[nt].u8[0] = *(const u16x8*)&lB[cur][col][hi * 8];
            bf[nt].u8[1] = *(const u16x8*)&lB[cur][col][16 + hi * 8];
        }
        #pragma unroll
        for (int mt = 0; mt < 2; ++mt)
            #pragma unroll
            for (int nt = 0; nt < 4; ++nt)
                acc[mt][nt] = wmma_bf16(af[mt].bf, bf[nt].bf, acc[mt][nt]);
        __syncthreads();
    }

    // ---- epilogue ----
    #pragma unroll
    for (int mt = 0; mt < 2; ++mt) {
        #pragma unroll
        for (int nt = 0; nt < 4; ++nt) {
            const int gc = n0 + wn * 64 + nt * 16 + lr;
            const float bb = bias[gc];
            #pragma unroll
            for (int i = 0; i < 8; ++i) {
                const int gr = m0 + wm * 32 + mt * 16 + hi * 8 + i;
                const float v = (acc[mt][nt][i] + bb) * scale;
                if constexpr (OUT_MODE == 0) {
                    ((float*)out)[(size_t)gr * Ndim + gc] = v;
                } else if constexpr (OUT_MODE == 1) {
                    ((u16*)out)[(size_t)gr * Ndim + gc] = f2bf(v);
                } else {
                    const int bI = gr >> 11;        // / Sc
                    const int s  = gr & (Sc - 1);
                    ((u16*)out)[(((size_t)(bI * 1024 + gc)) << 11) + s] = f2bf(v);
                }
            }
        }
    }
}

// ---------------- flash attention: 16 q-rows per wave, 64-wide kv chunks ----------------
// Q: [b*S+s][h*64+f] bf16 pre-scaled by 0.125*log2(e); K: same layout; Vt: [(b*1024+h*64+f)][s]
__global__ __launch_bounds__(256) void attn_kernel(const u16* __restrict__ Q,
                                                   const u16* __restrict__ K,
                                                   const u16* __restrict__ Vt,
                                                   u16* __restrict__ X) {
    __shared__ u16 lK[2][64][72];     // [kv][f], pitch 144B (TDM-padded)
    __shared__ u16 lV[2][64][72];     // [f][kv]
    __shared__ u16 lP[8][16][72];     // per-wave probs 16x64

    const int tid = threadIdx.x;
    const int wid = tid >> 5;
    const int lane = tid & 31;
    const int lr = lane & 15;
    const int hi = lane >> 4;

    const int bh = blockIdx.y;
    const int b = bh >> 4;
    const int h = bh & 15;
    const int q0 = blockIdx.x * 128 + wid * 16;

    const u16* Kbase = K  + (size_t)(b * Sc) * HDc + h * DHc;
    const u16* Vbase = Vt + ((size_t)(b * 1024 + h * DHc) << 11);   // * Sc

    Frag16 qa[2];
    {
        const size_t base = (size_t)(b * Sc + q0 + lr) * HDc + h * DHc;
        #pragma unroll
        for (int f = 0; f < 2; ++f) {
            qa[f].u8[0] = *(const u16x8*)(Q + base + f * 32 + hi * 8);
            qa[f].u8[1] = *(const u16x8*)(Q + base + f * 32 + 16 + hi * 8);
        }
    }

    float m_i[8], l_i[8];
    v8f accO[4] = {};
    #pragma unroll
    for (int i = 0; i < 8; ++i) { m_i[i] = -1e30f; l_i[i] = 0.0f; }

#ifdef HAVE_TDM
    if (wid == 0) {
        tdm_load_2d(Kbase, &lK[0][0][0], 64, 64, HDc, /*32 dw*/4, /*4 dw*/3);
        tdm_load_2d(Vbase, &lV[0][0][0], 64, 64, Sc, 4, 3);
    }
#endif

    for (int kv0 = 0; kv0 < Sc; kv0 += 64) {
        const int cur = (kv0 >> 6) & 1;
#ifdef HAVE_TDM
        if (wid == 0) {
            const int nxt = cur ^ 1;
            if (kv0 + 64 < Sc) {
                tdm_load_2d(Kbase + (size_t)(kv0 + 64) * HDc, &lK[nxt][0][0], 64, 64, HDc, 4, 3);
                tdm_load_2d(Vbase + (kv0 + 64),               &lV[nxt][0][0], 64, 64, Sc, 4, 3);
                __builtin_amdgcn_s_wait_tensorcnt((short)2);
            } else {
                __builtin_amdgcn_s_wait_tensorcnt((short)0);
            }
        }
#else
        {
            const int r = tid >> 2;                 // 0..63
            const int c0 = (tid & 3) * 16;          // 0..48
            const u16x8* ks = (const u16x8*)(Kbase + (size_t)(kv0 + r) * HDc + c0);
            *(u16x8*)&lK[cur][r][c0]     = ks[0];
            *(u16x8*)&lK[cur][r][c0 + 8] = ks[1];
            const u16x8* vs = (const u16x8*)(Vbase + (size_t)r * Sc + kv0 + c0);
            *(u16x8*)&lV[cur][r][c0]     = vs[0];
            *(u16x8*)&lV[cur][r][c0 + 8] = vs[1];
        }
#endif
        __syncthreads();

        // ---- scores: 16 q-rows x 64 kv-cols (8 wmma) ----
        v8f s[4];
        #pragma unroll
        for (int nt = 0; nt < 4; ++nt) {
            const int kvcol = nt * 16 + lr;
            Frag16 bk[2];
            #pragma unroll
            for (int f = 0; f < 2; ++f) {
                bk[f].u8[0] = *(const u16x8*)&lK[cur][kvcol][f * 32 + hi * 8];
                bk[f].u8[1] = *(const u16x8*)&lK[cur][kvcol][f * 32 + 16 + hi * 8];
            }
            v8f z = {};
            z = wmma_bf16(qa[0].bf, bk[0].bf, z);
            z = wmma_bf16(qa[1].bf, bk[1].bf, z);
            s[nt] = z;
        }

        // ---- online softmax in log2 domain ----
        #pragma unroll
        for (int i = 0; i < 8; ++i) {
            float lm = fmaxf(fmaxf(s[0][i], s[1][i]), fmaxf(s[2][i], s[3][i]));
            #pragma unroll
            for (int mm = 8; mm >= 1; mm >>= 1) lm = fmaxf(lm, __shfl_xor(lm, mm, 32));
            const float mn = fmaxf(m_i[i], lm);
            const float alpha = EXP2F(m_i[i] - mn);
            float p[4], rs = 0.0f;
            #pragma unroll
            for (int nt = 0; nt < 4; ++nt) { p[nt] = EXP2F(s[nt][i] - mn); rs += p[nt]; }
            #pragma unroll
            for (int mm = 8; mm >= 1; mm >>= 1) rs += __shfl_xor(rs, mm, 32);
            l_i[i] = l_i[i] * alpha + rs;
            m_i[i] = mn;
            #pragma unroll
            for (int ft = 0; ft < 4; ++ft) accO[ft][i] *= alpha;
            const int r = hi * 8 + i;
            #pragma unroll
            for (int nt = 0; nt < 4; ++nt) lP[wid][r][nt * 16 + lr] = f2bf(p[nt]);
        }
        __syncthreads();   // P (C-layout) -> LDS visible for A-layout reload

        // ---- O += P x V (8 wmma) ----
        Frag16 pa[2];
        #pragma unroll
        for (int f = 0; f < 2; ++f) {
            pa[f].u8[0] = *(const u16x8*)&lP[wid][lr][f * 32 + hi * 8];
            pa[f].u8[1] = *(const u16x8*)&lP[wid][lr][f * 32 + 16 + hi * 8];
        }
        #pragma unroll
        for (int ft = 0; ft < 4; ++ft) {
            const int fc = ft * 16 + lr;
            Frag16 b0, b1;
            b0.u8[0] = *(const u16x8*)&lV[cur][fc][hi * 8];
            b0.u8[1] = *(const u16x8*)&lV[cur][fc][16 + hi * 8];
            b1.u8[0] = *(const u16x8*)&lV[cur][fc][32 + hi * 8];
            b1.u8[1] = *(const u16x8*)&lV[cur][fc][48 + hi * 8];
            accO[ft] = wmma_bf16(pa[0].bf, b0.bf, accO[ft]);
            accO[ft] = wmma_bf16(pa[1].bf, b1.bf, accO[ft]);
        }
        __syncthreads();   // all waves done with cur buffers before TDM reuses them
    }

    // ---- normalize + store bf16 ----
    #pragma unroll
    for (int i = 0; i < 8; ++i) {
        const float inv = 1.0f / l_i[i];
        const size_t grow = (size_t)(b * Sc + q0 + hi * 8 + i) * HDc + h * DHc;
        #pragma unroll
        for (int ft = 0; ft < 4; ++ft)
            X[grow + ft * 16 + lr] = f2bf(accO[ft][i] * inv);
    }
}

// ---------------- host launch ----------------
extern "C" void kernel_launch(void* const* d_in, const int* in_sizes, int n_in,
                              void* d_out, int out_size, void* d_ws, size_t ws_size,
                              hipStream_t stream) {
    (void)in_sizes; (void)n_in; (void)out_size; (void)ws_size;
    const float* inq  = (const float*)d_in[0];
    const float* inkv = (const float*)d_in[1];
    const float* Wq = (const float*)d_in[2];
    const float* bq = (const float*)d_in[3];
    const float* Wk = (const float*)d_in[4];
    const float* bk = (const float*)d_in[5];
    const float* Wv = (const float*)d_in[6];
    const float* bv = (const float*)d_in[7];
    const float* Wo = (const float*)d_in[8];
    const float* bo = (const float*)d_in[9];
    float* out = (float*)d_out;

    const size_t MN = (size_t)Mrows * Dc;   // 8,388,608
    const size_t WN = (size_t)Dc * HDc;     // 1,048,576

    char* p = (char*)d_ws;
    u16* inq_bf  = (u16*)p; p += MN * 2;
    u16* inkv_bf = (u16*)p; p += MN * 2;
    u16* wqT = (u16*)p; p += WN * 2;
    u16* wkT = (u16*)p; p += WN * 2;
    u16* wvT = (u16*)p; p += WN * 2;
    u16* woT = (u16*)p; p += WN * 2;
    u16* qbf = (u16*)p; p += MN * 2;
    u16* kbf = (u16*)p; p += MN * 2;
    u16* vtbf = (u16*)p; p += MN * 2;
    u16* xbf  = (u16*)p; p += MN * 2;

    // stage 1: cast inputs, cast+transpose weights
    cast_f32_bf16<<<(int)(MN / 4 / 256), 256, 0, stream>>>(inq,  inq_bf,  (int)(MN / 4));
    cast_f32_bf16<<<(int)(MN / 4 / 256), 256, 0, stream>>>(inkv, inkv_bf, (int)(MN / 4));
    dim3 tgrid(32, 32);
    cast_transpose<<<tgrid, 256, 0, stream>>>(Wq, wqT);
    cast_transpose<<<tgrid, 256, 0, stream>>>(Wk, wkT);
    cast_transpose<<<tgrid, 256, 0, stream>>>(Wv, wvT);
    cast_transpose<<<tgrid, 256, 0, stream>>>(Wo, woT);

    // stage 2: QKV projections. Q folds 1/sqrt(DH) * log2(e) for exp2-domain softmax.
    const float qscale = 0.125f * 1.44269504088896340736f;
    dim3 ggrid(HDc / 128, Mrows / 128);   // (8, 64)
    gemm_kernel<1><<<ggrid, 256, 0, stream>>>(inq_bf,  wqT, bq, qbf,  qscale);
    gemm_kernel<1><<<ggrid, 256, 0, stream>>>(inkv_bf, wkT, bk, kbf,  1.0f);
    gemm_kernel<2><<<ggrid, 256, 0, stream>>>(inkv_bf, wvT, bv, vtbf, 1.0f);  // V^T layout

    // stage 3: flash attention
    dim3 agrid(Sc / 128, Bc * Hc);        // (16, 64)
    attn_kernel<<<agrid, 256, 0, stream>>>(qbf, kbf, vtbf, xbf);

    // stage 4: output projection, fp32 out
    gemm_kernel<0><<<ggrid, 256, 0, stream>>>(xbf, woT, bo, out, 1.0f);
}
